// ChunkedTriangleAttention_13168369729901
// MI455X (gfx1250) — matrix operand order repaired and checked by reading, not compiled, not run on hardware
//
#include <hip/hip_runtime.h>
#include <hip/hip_bf16.h>

// ---------------------------------------------------------------------------
// Triangle attention on MI455X (gfx1250, wave32, WMMA).
//
//   k_prep : z = sum_rank(z_left)+sum_rank(z_right); LayerNorm -> zn (bf16)
//   k_pack : Wq/Wk/Wv/Wout/Wgate -> bf16, pre-swizzled into the WMMA
//            B-fragment-friendly [n][k/2]-dword-pair layout
//   k_qkv  : zn @ [Wq|Wk|Wv] + bias -> q[h][l][d], k[h][l][d], vT[h][d][l]
//   k_attn : flash attention per (head, 64-query tile); K/V tiles staged to
//            LDS once per WG via global_load_async_to_lds_b128 (ASYNCcnt),
//            QK^T and PV via v_wmma_f32_16x16x32_bf16, online softmax,
//            P relayout through LDS
//   k_final: attn_out @ Wout + bout, gated by sigmoid(zn @ Wgate + bgate),
//            /RANK broadcast into out_left
//   k_zero : out_right = 0
//
// Per-query attention bias (z_left_sum @ Wbias) is constant across keys ->
// exact softmax no-op; omitted. Key mask bias INF*(mask-1) is applied.
// ---------------------------------------------------------------------------

#define LSEQ 2048
#define CP   128
#define CH   512
#define NH   8
#define HD   64

typedef __attribute__((ext_vector_type(16))) __bf16 v16bf;
typedef __attribute__((ext_vector_type(8)))  float  v8f;
typedef __attribute__((ext_vector_type(4)))  int    i4;

union FragU { v16bf v; i4 q[2]; };

__device__ __forceinline__ unsigned short f2bf(float f) {
    unsigned int u = __float_as_uint(f);
    u = (u + 0x7FFFu + ((u >> 16) & 1u)) >> 16;     // round-to-nearest-even
    return (unsigned short)u;
}

__device__ __forceinline__ v8f wmma_bf16(v16bf a, v16bf b, v8f c) {
    return __builtin_amdgcn_wmma_f32_16x16x32_bf16(
        false, a, false, b, (short)0, c, false, false);
}

// gfx1250 async copy global->LDS, 16B per lane (tracked by ASYNCcnt).
// LDS destination address = low 32 bits of the generic shared-mem address
// (generic LDS aperture maps addr[31:0] directly to the LDS offset).
__device__ __forceinline__ void async_g2l_b128(const void* gptr, void* lptr) {
    unsigned lds = (unsigned)(unsigned long long)lptr;
    asm volatile("global_load_async_to_lds_b128 %0, %1, off"
                 :: "v"(lds), "v"(gptr) : "memory");
}
__device__ __forceinline__ void wait_asynccnt0() {
    asm volatile("s_wait_asynccnt 0" ::: "memory");
}

// ------------------------- workspace layout (bytes) ------------------------
#define OFF_ZN   (size_t)0                 // 2048*128 bf16      = 524288
#define OFF_Q    (size_t)524288            // 8*2048*64 bf16     = 2097152
#define OFF_K    (size_t)2621440
#define OFF_V    (size_t)4718592           // vT: [h][d][l]
#define OFF_AO   (size_t)6815744           // 2048*512 bf16      = 2097152
#define OFF_WQP  (size_t)8912896           // 512*64 dwords      = 131072
#define OFF_WKP  (size_t)9043968
#define OFF_WVP  (size_t)9175040
#define OFF_WOP  (size_t)9306112           // 128*256 dwords     = 131072
#define OFF_WGP  (size_t)9437184           // 128*64 dwords      = 32768

// --------------------------- kernel: prep + LN -----------------------------
__global__ void k_prep(const float* __restrict__ zl, const float* __restrict__ zr,
                       const float* __restrict__ ln_g, const float* __restrict__ ln_b,
                       unsigned short* __restrict__ zn) {
    int l = blockIdx.x, c = threadIdx.x;            // 2048 blocks x 128 threads
    const float* pl = zl + (size_t)l * 4 * CP + c;
    const float* pr = zr + (size_t)l * 4 * CP + c;
    float z = 0.f;
#pragma unroll
    for (int r = 0; r < 4; ++r) z += pl[r * CP] + pr[r * CP];

    __shared__ float red[CP];
    red[c] = z; __syncthreads();
#pragma unroll
    for (int off = 64; off > 0; off >>= 1) {
        if (c < off) red[c] += red[c + off];
        __syncthreads();
    }
    float mu = red[0] * (1.0f / CP);
    __syncthreads();
    float d = z - mu;
    red[c] = d * d; __syncthreads();
#pragma unroll
    for (int off = 64; off > 0; off >>= 1) {
        if (c < off) red[c] += red[c + off];
        __syncthreads();
    }
    float var = red[0] * (1.0f / CP);
    float v = d * __frsqrt_rn(var + 1e-5f) * ln_g[c] + ln_b[c];
    zn[l * CP + c] = f2bf(v);
}

// -------------------- kernel: pack weights into B layout -------------------
// B-fragment layout: dword[n][k2] = { bf16(W[2*k2][n]), bf16(W[2*k2+1][n]) }
__global__ void k_pack(const float* __restrict__ Wq, const float* __restrict__ Wk,
                       const float* __restrict__ Wv, const float* __restrict__ Wo,
                       const float* __restrict__ Wg,
                       unsigned int* __restrict__ WqP, unsigned int* __restrict__ WkP,
                       unsigned int* __restrict__ WvP, unsigned int* __restrict__ WoP,
                       unsigned int* __restrict__ WgP) {
    int t = blockIdx.x * 256 + threadIdx.x;
    if (t < 32768) {                                   // Wq: [128][512], K=128
        int n = t >> 6, k2 = t & 63;
        WqP[n * 64 + k2] = (unsigned)f2bf(Wq[(2 * k2) * CH + n]) |
                           ((unsigned)f2bf(Wq[(2 * k2 + 1) * CH + n]) << 16);
    } else if (t < 65536) {
        int u = t - 32768, n = u >> 6, k2 = u & 63;
        WkP[n * 64 + k2] = (unsigned)f2bf(Wk[(2 * k2) * CH + n]) |
                           ((unsigned)f2bf(Wk[(2 * k2 + 1) * CH + n]) << 16);
    } else if (t < 98304) {
        int u = t - 65536, n = u >> 6, k2 = u & 63;
        WvP[n * 64 + k2] = (unsigned)f2bf(Wv[(2 * k2) * CH + n]) |
                           ((unsigned)f2bf(Wv[(2 * k2 + 1) * CH + n]) << 16);
    } else if (t < 131072) {                           // Wout: [512][128], K=512
        int u = t - 98304, n = u >> 8, k2 = u & 255;
        WoP[n * 256 + k2] = (unsigned)f2bf(Wo[(2 * k2) * CP + n]) |
                            ((unsigned)f2bf(Wo[(2 * k2 + 1) * CP + n]) << 16);
    } else if (t < 139264) {                           // Wgate: [128][128], K=128
        int u = t - 131072, n = u >> 6, k2 = u & 63;
        WgP[n * 64 + k2] = (unsigned)f2bf(Wg[(2 * k2) * CP + n]) |
                           ((unsigned)f2bf(Wg[(2 * k2 + 1) * CP + n]) << 16);
    }
}

// --------------------------- kernel: QKV GEMM ------------------------------
// [2048 x 128] @ [128 x 512] x3. Grid (12, 32): x = 128-wide N tile over
// concat(q|k|v); y = 64-row M tile. 4 waves, each owns 16 rows x 128 cols.
__global__ void k_qkv(const unsigned short* __restrict__ zn,
                      const unsigned int* __restrict__ WqP,
                      const unsigned int* __restrict__ WkP,
                      const unsigned int* __restrict__ WvP,
                      const float* __restrict__ bq, const float* __restrict__ bk,
                      const float* __restrict__ bv,
                      unsigned short* __restrict__ qbuf,
                      unsigned short* __restrict__ kbuf,
                      unsigned short* __restrict__ vT) {
    int wave = threadIdx.x >> 5, lane = threadIdx.x & 31;
    int half = lane >> 4, ln = lane & 15;
    int m0 = blockIdx.y * 64 + wave * 16;
    int n0 = blockIdx.x * 128;
    int seg = n0 / CH, nseg = n0 % CH;
    const unsigned int* WP = (seg == 0) ? WqP : (seg == 1) ? WkP : WvP;
    const float* bias = (seg == 0) ? bq : (seg == 1) ? bk : bv;

    v8f c[8] = {};
#pragma unroll
    for (int ks = 0; ks < 4; ++ks) {
        int k0 = ks * 32;
        FragU a;
        const unsigned short* ap = zn + (m0 + ln) * CP + k0 + half * 8;
        a.q[0] = *(const i4*)ap;
        a.q[1] = *(const i4*)(ap + 16);
#pragma unroll
        for (int ns = 0; ns < 8; ++ns) {
            int n = nseg + ns * 16 + ln;
            const unsigned int* bp = WP + n * 64 + ks * 16 + half * 4;
            FragU b;
            b.q[0] = *(const i4*)bp;
            b.q[1] = *(const i4*)(bp + 8);
            c[ns] = wmma_bf16(a.v, b.v, c[ns]);
        }
    }
#pragma unroll
    for (int ns = 0; ns < 8; ++ns) {
        int n_in = nseg + ns * 16 + ln;
        float bb = bias[n_in];
        int h = n_in >> 6, d = n_in & 63;
#pragma unroll
        for (int r = 0; r < 8; ++r) {
            int row = m0 + r + 8 * half;
            unsigned short hb = f2bf(c[ns][r] + bb);
            if (seg == 0)      qbuf[(h * LSEQ + row) * HD + d] = hb;
            else if (seg == 1) kbuf[(h * LSEQ + row) * HD + d] = hb;
            else               vT[(h * HD + d) * LSEQ + row] = hb;
        }
    }
}

// -------------------------- kernel: flash attention ------------------------
// Grid 256: block = (head, 64-query tile). 4 waves x 16 query rows.
// K/V tiles staged to LDS once per workgroup via async global->LDS copies.
__global__ void k_attn(const unsigned short* __restrict__ qbuf,
                       const unsigned short* __restrict__ kbuf,
                       const unsigned short* __restrict__ vT,
                       const float* __restrict__ mask,
                       unsigned short* __restrict__ attnout) {
    int head = blockIdx.x >> 5, qt = blockIdx.x & 31;
    int wave = threadIdx.x >> 5, lane = threadIdx.x & 31;
    int half = lane >> 4, ln = lane & 15;
    int tid  = threadIdx.x;
    int q0 = qt * 64 + wave * 16;

    __shared__ unsigned short ldsK[64 * 64];          // K tile  [key][d],  8KB
    __shared__ unsigned short ldsV[64 * 64];          // V tile  [d][key],  8KB
    __shared__ unsigned short pT[4 * 16 * 64];        // per-wave P tiles,  8KB
    unsigned short* myP = pT + wave * 1024;

    // Q A-fragments (held for whole kernel)
    const unsigned short* qp = qbuf + ((size_t)head * LSEQ + q0 + ln) * HD;
    FragU aQ0, aQ1;
    aQ0.q[0] = *(const i4*)(qp + half * 8);       aQ0.q[1] = *(const i4*)(qp + 16 + half * 8);
    aQ1.q[0] = *(const i4*)(qp + 32 + half * 8);  aQ1.q[1] = *(const i4*)(qp + 48 + half * 8);

    float mrun[8], lrun[8];
#pragma unroll
    for (int r = 0; r < 8; ++r) { mrun[r] = -1e30f; lrun[r] = 0.f; }
    v8f o[4] = {};
    const float scale = 0.125f;                       // 1/sqrt(64)

    const unsigned short* kBase = kbuf + (size_t)head * LSEQ * HD;
    const unsigned short* vBase = vT + (size_t)head * HD * LSEQ;

    for (int kt = 0; kt < 32; ++kt) {
        int kb = kt * 64;

        // ---- stage K/V tile into LDS (128 lanes x 4 chunks x 16B each) ----
        __syncthreads();                              // prev tile reads done
#pragma unroll
        for (int i = 0; i < 4; ++i) {
            int chunk = tid + 128 * i;                // 0..511
            int row = chunk >> 3, col8 = (chunk & 7) * 8;
            async_g2l_b128(kBase + (kb + row) * HD + col8, ldsK + row * 64 + col8);
            async_g2l_b128(vBase + row * LSEQ + kb + col8, ldsV + row * 64 + col8);
        }
        wait_asynccnt0();
        __syncthreads();                              // all waves see the tile

        // ---- S = (Q K^T) * scale + mask-bias ----
        v8f s[4];
#pragma unroll
        for (int ns = 0; ns < 4; ++ns) {
            int keyl = ns * 16 + ln;                  // key row within tile
            const unsigned short* kp = ldsK + keyl * 64;
            FragU b0, b1;
            b0.q[0] = *(const i4*)(kp + half * 8);      b0.q[1] = *(const i4*)(kp + half * 8 + 16);
            b1.q[0] = *(const i4*)(kp + 32 + half * 8); b1.q[1] = *(const i4*)(kp + 32 + half * 8 + 16);
            v8f cc = {};
            cc = wmma_bf16(aQ0.v, b0.v, cc);
            cc = wmma_bf16(aQ1.v, b1.v, cc);
            float mb = 1000000000.0f * (mask[kb + keyl] - 1.0f);
#pragma unroll
            for (int r = 0; r < 8; ++r) s[ns][r] = cc[r] * scale + mb;
        }

        // ---- online softmax per row (rows split across lane halves) ----
        float corr[8];
#pragma unroll
        for (int r = 0; r < 8; ++r) {
            float mx = fmaxf(fmaxf(s[0][r], s[1][r]), fmaxf(s[2][r], s[3][r]));
#pragma unroll
            for (int off = 8; off > 0; off >>= 1)
                mx = fmaxf(mx, __shfl_xor(mx, off, 32));
            float mnew = fmaxf(mrun[r], mx);
            float co = __expf(mrun[r] - mnew);
            float rs = 0.f;
#pragma unroll
            for (int ns = 0; ns < 4; ++ns) {
                float p = __expf(s[ns][r] - mnew);
                s[ns][r] = p; rs += p;
            }
#pragma unroll
            for (int off = 8; off > 0; off >>= 1)
                rs += __shfl_xor(rs, off, 32);
            lrun[r] = lrun[r] * co + rs;
            mrun[r] = mnew;
            corr[r] = co;
        }
#pragma unroll
        for (int d4 = 0; d4 < 4; ++d4)
#pragma unroll
            for (int r = 0; r < 8; ++r) o[d4][r] *= corr[r];

        // ---- P (C-layout) -> LDS -> A-layout fragments ----
#pragma unroll
        for (int ns = 0; ns < 4; ++ns)
#pragma unroll
            for (int r = 0; r < 8; ++r)
                myP[(r + 8 * half) * 64 + ns * 16 + ln] = f2bf(s[ns][r]);
        asm volatile("s_wait_dscnt 0" ::: "memory");   // DS in-order per wave
        const unsigned short* pp = myP + ln * 64;
        FragU pa0, pa1;
        pa0.q[0] = *(const i4*)(pp + half * 8);       pa0.q[1] = *(const i4*)(pp + 16 + half * 8);
        pa1.q[0] = *(const i4*)(pp + 32 + half * 8);  pa1.q[1] = *(const i4*)(pp + 48 + half * 8);

        // ---- O += P V ----
#pragma unroll
        for (int d4 = 0; d4 < 4; ++d4) {
            int n = d4 * 16 + ln;                     // d row within V tile
            const unsigned short* vp = ldsV + n * 64;
            FragU bv0, bv1;
            bv0.q[0] = *(const i4*)(vp + half * 8);      bv0.q[1] = *(const i4*)(vp + half * 8 + 16);
            bv1.q[0] = *(const i4*)(vp + 32 + half * 8); bv1.q[1] = *(const i4*)(vp + 32 + half * 8 + 16);
            o[d4] = wmma_bf16(pa0.v, bv0.v, o[d4]);
            o[d4] = wmma_bf16(pa1.v, bv1.v, o[d4]);
        }
    }
    // finalize and store attn output (row-major [L][CH] bf16)
#pragma unroll
    for (int d4 = 0; d4 < 4; ++d4)
#pragma unroll
        for (int r = 0; r < 8; ++r) {
            float val = o[d4][r] / lrun[r];
            int row = q0 + r + 8 * half;
            attnout[row * CH + head * HD + d4 * 16 + ln] = f2bf(val);
        }
}

// ------------------ kernel: Wout GEMM + gate + broadcast out ---------------
__global__ void k_final(const unsigned short* __restrict__ attnout,
                        const unsigned short* __restrict__ zn,
                        const unsigned int* __restrict__ WoP,
                        const unsigned int* __restrict__ WgP,
                        const float* __restrict__ bout,
                        const float* __restrict__ bgate,
                        float* __restrict__ out_left) {
    int wave = threadIdx.x >> 5, lane = threadIdx.x & 31;
    int half = lane >> 4, ln = lane & 15;
    int m0 = blockIdx.x * 64 + wave * 16;

    // out = attnout[2048,512] @ Wout[512,128]
    v8f c[8] = {};
#pragma unroll
    for (int ks = 0; ks < 16; ++ks) {
        FragU a;
        const unsigned short* ap = attnout + (m0 + ln) * CH + ks * 32 + half * 8;
        a.q[0] = *(const i4*)ap;
        a.q[1] = *(const i4*)(ap + 16);
#pragma unroll
        for (int ns = 0; ns < 8; ++ns) {
            int n = ns * 16 + ln;
            const unsigned int* bp = WoP + n * 256 + ks * 16 + half * 4;
            FragU b;
            b.q[0] = *(const i4*)bp;
            b.q[1] = *(const i4*)(bp + 8);
            c[ns] = wmma_bf16(a.v, b.v, c[ns]);
        }
    }
    // gate = zn[2048,128] @ Wgate[128,128]
    v8f g[8] = {};
#pragma unroll
    for (int ks = 0; ks < 4; ++ks) {
        FragU a;
        const unsigned short* ap = zn + (m0 + ln) * CP + ks * 32 + half * 8;
        a.q[0] = *(const i4*)ap;
        a.q[1] = *(const i4*)(ap + 16);
#pragma unroll
        for (int ns = 0; ns < 8; ++ns) {
            int n = ns * 16 + ln;
            const unsigned int* bp = WgP + n * 64 + ks * 16 + half * 4;
            FragU b;
            b.q[0] = *(const i4*)bp;
            b.q[1] = *(const i4*)(bp + 8);
            g[ns] = wmma_bf16(a.v, b.v, g[ns]);
        }
    }
#pragma unroll
    for (int ns = 0; ns < 8; ++ns) {
        int n = ns * 16 + ln;
        float bo = bout[n], bg = bgate[n];
#pragma unroll
        for (int r = 0; r < 8; ++r) {
            int row = m0 + r + 8 * half;
            float ov = c[ns][r] + bo;
            float gv = 1.0f / (1.0f + __expf(-(g[ns][r] + bg)));
            float fin = ov * gv * 0.25f;               // / RANK
            float* dst = out_left + ((size_t)row * 4) * CP + n;
            dst[0]       = fin;
            dst[CP]      = fin;
            dst[2 * CP]  = fin;
            dst[3 * CP]  = fin;
        }
    }
}

// ---------------------------- kernel: zero out_right -----------------------
__global__ void k_zero(i4* __restrict__ p, int n4) {
    int t = blockIdx.x * 256 + threadIdx.x;
    if (t < n4) { i4 z = {0, 0, 0, 0}; p[t] = z; }
}

// ---------------------------------------------------------------------------
extern "C" void kernel_launch(void* const* d_in, const int* in_sizes, int n_in,
                              void* d_out, int out_size, void* d_ws, size_t ws_size,
                              hipStream_t stream) {
    const float* zl    = (const float*)d_in[0];
    const float* zr    = (const float*)d_in[1];
    const float* mask  = (const float*)d_in[2];
    const float* ln_g  = (const float*)d_in[3];
    const float* ln_b  = (const float*)d_in[4];
    const float* Wq    = (const float*)d_in[5];
    const float* bq    = (const float*)d_in[6];
    const float* Wk    = (const float*)d_in[7];
    const float* bk    = (const float*)d_in[8];
    const float* Wv    = (const float*)d_in[9];
    const float* bv    = (const float*)d_in[10];
    // d_in[11] = Wbias: per-query bias is constant across keys -> softmax no-op
    const float* Wo    = (const float*)d_in[12];
    const float* bo    = (const float*)d_in[13];
    const float* Wg    = (const float*)d_in[14];
    const float* bg    = (const float*)d_in[15];

    char* w = (char*)d_ws;
    unsigned short* zn   = (unsigned short*)(w + OFF_ZN);
    unsigned short* qbuf = (unsigned short*)(w + OFF_Q);
    unsigned short* kbuf = (unsigned short*)(w + OFF_K);
    unsigned short* vT   = (unsigned short*)(w + OFF_V);
    unsigned short* ao   = (unsigned short*)(w + OFF_AO);
    unsigned int*   WqP  = (unsigned int*)(w + OFF_WQP);
    unsigned int*   WkP  = (unsigned int*)(w + OFF_WKP);
    unsigned int*   WvP  = (unsigned int*)(w + OFF_WVP);
    unsigned int*   WoP  = (unsigned int*)(w + OFF_WOP);
    unsigned int*   WgP  = (unsigned int*)(w + OFF_WGP);

    float* out = (float*)d_out;                       // [out_left | out_right]
    const int N_LEFT = LSEQ * 4 * CP;                 // 1048576 floats each

    k_prep<<<LSEQ, 128, 0, stream>>>(zl, zr, ln_g, ln_b, zn);
    k_pack<<<(139264 + 255) / 256, 256, 0, stream>>>(Wq, Wk, Wv, Wo, Wg,
                                                     WqP, WkP, WvP, WoP, WgP);
    k_qkv<<<dim3(12, 32), 128, 0, stream>>>(zn, WqP, WkP, WvP, bq, bk, bv,
                                            qbuf, kbuf, vT);
    k_attn<<<NH * 32, 128, 0, stream>>>(qbuf, kbuf, vT, mask, ao);
    k_final<<<LSEQ / 64, 128, 0, stream>>>(ao, zn, WoP, WgP, bo, bg, out);
    k_zero<<<(N_LEFT / 4 + 255) / 256, 256, 0, stream>>>(
        (i4*)(out + N_LEFT), N_LEFT / 4);
}
